// QHFTLinear_84456236909364
// MI455X (gfx1250) — matrix operand order repaired and checked by reading, not compiled, not run on hardware
//
#include <hip/hip_runtime.h>

#define OUT_F  4096
#define IN_F   4096
#define NFREQ  10000
#define TOKENS 1024   // 2 * 512

typedef __attribute__((ext_vector_type(16))) __bf16 v16bf;
typedef __attribute__((ext_vector_type(8)))  __bf16 v8bf;
typedef __attribute__((ext_vector_type(8)))  float  v8f;

// ---------------------------------------------------------------- zero output
__global__ void qh_zero_f32(float* __restrict__ p, int n) {
    int i = blockIdx.x * blockDim.x + threadIdx.x;
    if (i < n) p[i] = 0.0f;
}

// -------------------------------------------------- scatter-set dedup (last j wins)
// alive[j] = 1 iff no later j' has the same (row, col) pair.
__global__ void qh_dedup(const int* __restrict__ idx, int* __restrict__ alive) {
    const int j = blockIdx.x;
    const int r = idx[j];
    const int c = idx[NFREQ + j];
    __shared__ int found;
    if (threadIdx.x == 0) found = 0;
    __syncthreads();
    for (int jp = j + 1 + threadIdx.x; jp < NFREQ; jp += blockDim.x) {
        if (idx[jp] == r && idx[NFREQ + jp] == c) { found = 1; break; }
    }
    __syncthreads();
    if (threadIdx.x == 0) alive[j] = !found;
}

// ------------------------------------- y[t, r_j] += spectrum[j] * x[t, c_j]  (fp32)
__global__ void qh_scatter(const int* __restrict__ idx, const float* __restrict__ spec,
                           const int* __restrict__ alive, const float* __restrict__ x,
                           float* __restrict__ z) {
    const int j = blockIdx.x;
    if (!alive[j]) return;
    const int   r = idx[j];          // out index (FWHT axis)
    const int   c = idx[NFREQ + j];  // in index
    const float v = spec[j];
    for (int t = threadIdx.x; t < TOKENS; t += blockDim.x)
        atomicAdd(&z[(size_t)t * OUT_F + r], v * x[(size_t)t * IN_F + c]);
}

// ---------------------- in-place FWHT over each length-4096 row, then scale by s
__global__ void qh_fwht(float* __restrict__ z) {
    __shared__ float buf[OUT_F];
    float* row = z + (size_t)blockIdx.x * OUT_F;
    for (int e = threadIdx.x; e < OUT_F; e += blockDim.x) buf[e] = row[e];
    __syncthreads();
    for (int h = 1; h < OUT_F; h <<= 1) {
        for (int p = threadIdx.x; p < OUT_F / 2; p += blockDim.x) {
            int lo = ((p & ~(h - 1)) << 1) | (p & (h - 1));
            int hi = lo + h;
            float a = buf[lo], b = buf[hi];
            buf[lo] = a + b;
            buf[hi] = a - b;
        }
        __syncthreads();
    }
    const float s = 150.0f / 64.0f;   // SCALING / sqrt(OUT_F)
    for (int e = threadIdx.x; e < OUT_F; e += blockDim.x) row[e] = buf[e] * s;
}

// ----------------------------------------------------------- f32 -> bf16 convert
__global__ void qh_cvt_bf16(const float* __restrict__ src, __bf16* __restrict__ dst, int n) {
    int i = blockIdx.x * blockDim.x + threadIdx.x;
    if (i < n) dst[i] = (__bf16)src[i];
}

// --------------------------------------------------------------- WMMA bf16 GEMM
// out[m, n] += sum_k Xb[m, k] * Wb[n, k];  M=1024, N=4096, K=4096
// Block = 128 threads = 4 waves; each wave computes a 64x64 tile (4x4 fragments).
// A frag (16x32, M x K, row-major): lane l -> row m0+16i+(l&15),
//   elems 0..15 = K[k0 + 16*(l>>4) .. +15]
// B frag (32x16, K x N) with W row-major [N, K]: lane l -> row n0+16j+(l&15),
//   elems 0..15 = K[k0 + 16*(l>>4) .. +15]  (one contiguous 32B load)
__global__ __launch_bounds__(128)
void qh_gemm(const __bf16* __restrict__ Xb, const __bf16* __restrict__ Wb,
             float* __restrict__ out) {
    const int lane   = threadIdx.x & 31;
    const int wave   = threadIdx.x >> 5;
    const int laneLo = lane & 15;
    const int laneHi = lane >> 4;
    const int m0 = blockIdx.x * 64;
    const int n0 = blockIdx.y * 256 + wave * 64;

    v8f acc[4][4] = {};

    for (int k0 = 0; k0 < IN_F; k0 += 32) {
        v16bf a[4], b[4];
#pragma unroll
        for (int i = 0; i < 4; ++i) {
            const __bf16* pa = Xb + (size_t)(m0 + 16 * i + laneLo) * IN_F + k0 + laneHi * 16;
            v8bf lo = *(const v8bf*)pa;
            v8bf hi = *(const v8bf*)(pa + 8);
            a[i] = __builtin_shufflevector(lo, hi,
                    0, 1, 2, 3, 4, 5, 6, 7, 8, 9, 10, 11, 12, 13, 14, 15);
        }
#pragma unroll
        for (int j = 0; j < 4; ++j) {
            const __bf16* pb = Wb + (size_t)(n0 + 16 * j + laneLo) * IN_F + k0 + laneHi * 16;
            b[j] = *(const v16bf*)pb;
        }
#pragma unroll
        for (int i = 0; i < 4; ++i)
#pragma unroll
            for (int j = 0; j < 4; ++j)
                acc[i][j] = __builtin_amdgcn_wmma_f32_16x16x32_bf16(
                    false, a[i], false, b[j], (short)0, acc[i][j], false, false);
    }

    // C/D layout: element (M = 8*laneHi + v, N = laneLo) in 16x16 tile, VGPR v.
#pragma unroll
    for (int i = 0; i < 4; ++i) {
#pragma unroll
        for (int j = 0; j < 4; ++j) {
            const int col = n0 + 16 * j + laneLo;
#pragma unroll
            for (int v = 0; v < 8; ++v) {
                const int rowm = m0 + 16 * i + laneHi * 8 + v;
                out[(size_t)rowm * OUT_F + col] += acc[i][j][v];
            }
        }
    }
}

extern "C" void kernel_launch(void* const* d_in, const int* in_sizes, int n_in,
                              void* d_out, int out_size, void* d_ws, size_t ws_size,
                              hipStream_t stream) {
    const float* x    = (const float*)d_in[0];   // [2,512,4096] -> [1024,4096]
    const float* w    = (const float*)d_in[1];   // [4096,4096] row-major (O, I)
    const float* spec = (const float*)d_in[2];   // [10000]
    const int*   idx  = (const int*)d_in[3];     // [2,10000]
    float* out = (float*)d_out;                  // [1024,4096]

    char*   ws    = (char*)d_ws;
    __bf16* Wb    = (__bf16*)ws;                                         // 32 MB
    __bf16* Xb    = (__bf16*)(ws + (size_t)OUT_F * IN_F * 2);            //  8 MB
    int*    alive = (int*)(ws + (size_t)OUT_F * IN_F * 2
                               + (size_t)TOKENS * IN_F * 2);             // 40 KB

    // 1) out = 0
    qh_zero_f32<<<(TOKENS * OUT_F) / 256, 256, 0, stream>>>(out, TOKENS * OUT_F);
    // 2) scatter-set dedup (last occurrence wins, sequential semantics)
    qh_dedup<<<NFREQ, 256, 0, stream>>>(idx, alive);
    // 3) out[t, r_j] += s_j * x[t, c_j]   (exact fp32 sparse path)
    qh_scatter<<<NFREQ, 256, 0, stream>>>(idx, spec, alive, x, out);
    // 4) out rows <- FWHT(out rows) * (150/64)
    qh_fwht<<<TOKENS, 256, 0, stream>>>(out);
    // 5) bf16 copies of W and x for the WMMA GEMM
    qh_cvt_bf16<<<(OUT_F * IN_F) / 256, 256, 0, stream>>>(w, Wb, OUT_F * IN_F);
    qh_cvt_bf16<<<(TOKENS * IN_F) / 256, 256, 0, stream>>>(x, Xb, TOKENS * IN_F);
    // 6) out += x @ W^T  via v_wmma_f32_16x16x32_bf16
    dim3 grid(TOKENS / 64, OUT_F / 256);
    qh_gemm<<<grid, 128, 0, stream>>>(Xb, Wb, out);
}
